// CliffordEquivariantGatingBlock_2774548873278
// MI455X (gfx1250) — matrix operand order repaired and checked by reading, not compiled, not run on hardware
//
#include <hip/hip_runtime.h>
#include <stdint.h>

// ---------------------------------------------------------------------------
// CDNA5 / gfx1250 fused Clifford equivariant gating block.
//   probe/match GEMMs via V_WMMA_F32_16X16X32_BF16 with fp32 emulated as
//   bf16 hi+lo (3 WMMAs per fp32 MAC tile), fused packed-fp32 epilogue.
// ---------------------------------------------------------------------------

typedef __bf16   bf16_t;
typedef bf16_t   v16bf __attribute__((ext_vector_type(16)));
typedef float    v8f   __attribute__((ext_vector_type(8)));
typedef float    v4f   __attribute__((ext_vector_type(4)));
typedef float    v2f   __attribute__((ext_vector_type(2)));
typedef uint32_t v4u   __attribute__((ext_vector_type(4)));

__device__ __forceinline__ uint32_t f2bf_rne(float f) {
  uint32_t u = __float_as_uint(f);
  u += 0x7FFFu + ((u >> 16) & 1u);   // round-to-nearest-even to bf16
  return u >> 16;
}
__device__ __forceinline__ float bf2f(uint32_t h) {
  return __uint_as_float(h << 16);
}
__device__ __forceinline__ uint32_t pack2(uint32_t lo16, uint32_t hi16) {
  return (lo16 & 0xFFFFu) | (hi16 << 16);
}

// ---------------------------------------------------------------------------
// Pre-kernel: convert w_probe/w_match (OUT=64, IN=64, 4 channels) fp32 into
// bf16 hi/lo B-matrix fragments (32x16, K x N) in a contiguous-K VGPR layout (same K order as the A staging):
//   VGPR v, lanes 0-15:  N=lane,    K = kc*32 + 2v, 2v+1   (even K in [15:0])
//   VGPR v, lanes 16-31: N=lane-16, K = kc*32 + 16 + 2v, +1
// ws layout (dwords): [s(2)][c(4)][kc(2)][nb(4)][hl(2)][lane(32)][v(8)]
//   = 32768 dwords = 128 KB in d_ws.
// ---------------------------------------------------------------------------
__global__ void clifford_prep_w(const float* __restrict__ wp,
                                const float* __restrict__ wm,
                                uint32_t* __restrict__ ws) {
  int g = blockIdx.x * blockDim.x + threadIdx.x;   // 0..32767
  int v    = g & 7;
  int lane = (g >> 3) & 31;
  int hl   = (g >> 8) & 1;
  int nb   = (g >> 9) & 3;
  int kc   = (g >> 11) & 1;
  int c    = (g >> 12) & 3;
  int s    = (g >> 14) & 1;
  int n = nb * 16 + (lane & 15);
  int K = kc * 32 + (lane >> 4) * 16 + 2 * v;
  const float* w = s ? wm : wp;
  float f0 = w[n * 256 + K * 4 + c];           // W[n][k][c]
  float f1 = w[n * 256 + (K + 1) * 4 + c];
  uint32_t h0 = f2bf_rne(f0), h1 = f2bf_rne(f1);
  uint32_t d;
  if (hl == 0) d = pack2(h0, h1);                                          // hi
  else         d = pack2(f2bf_rne(f0 - bf2f(h0)), f2bf_rne(f1 - bf2f(h1))); // lo
  ws[g] = d;
}

// ---------------------------------------------------------------------------
// Main fused kernel: 256 threads = 8 waves, 32 batch rows per workgroup.
// Wave w: mb = w&1 (16-row block), nb = w>>1 (16-col block).
// __launch_bounds__(256, 4): cap at 256 VGPR so 2 workgroups share a WGP and
// one WG's HBM streaming overlaps the other's WMMA/epilogue phases.
// ---------------------------------------------------------------------------
__launch_bounds__(256, 4)
__global__ void clifford_main(const float* __restrict__ x,
                              const uint32_t* __restrict__ wsB,
                              const float* __restrict__ bp_,
                              const float* __restrict__ bm_,
                              float* __restrict__ out) {
  // A-fragments: [mb(2)][blade(8)][kc(2)][hl(2)][lane(32)][v(8)] dwords = 64 KB
  __shared__ uint32_t Afr[16384];

  const int tid  = threadIdx.x;
  const int lane = tid & 31;
  const int wv   = tid >> 5;
  const long rowBase = (long)blockIdx.x * 32;
  const float* xt = x + rowBase * 512;

  // ---- Stage x tile: fp32 -> bf16 hi/lo, directly in WMMA A-fragment layout
  // A (16x32 bf16) layout: lanes 0-15 M=lane, lanes 16-31 M=lane-16;
//   v0-7: K = (lane<16 ? 0..15 : 16..31)
#pragma unroll
  for (int it = 0; it < 8; ++it) {
    int flat = it * 256 + tid;       // 0..2047
    int iq = flat & 1;               // blade quad
    int kp = (flat >> 1) & 31;       // K pair (K = 2*kp, 2*kp+1)
    int m  = flat >> 6;              // 0..31
    const float* src = xt + m * 512 + kp * 16 + iq * 4;
    v4f fe = __builtin_nontemporal_load((const v4f*)src);       // K even
    v4f fo = __builtin_nontemporal_load((const v4f*)(src + 8)); // K odd
    int kc   = kp >> 4;
    int kk   = (2 * kp) & 31;
    int half = kk >> 4;
    int v    = (kk & 15) >> 1;
    int ld   = (m & 15) + half * 16;       // destination lane
    int mb   = m >> 4;
#pragma unroll
    for (int j = 0; j < 4; ++j) {
      int blade = iq * 4 + j;
      // hi = bf16(round-half-up); lo = bf16(f - hi). Residual error ~2^-16,
      // so half-up (1 add) instead of RNE (3 ops) is numerically irrelevant.
      uint32_t re = __float_as_uint(fe[j]) + 0x8000u;
      uint32_t ro = __float_as_uint(fo[j]) + 0x8000u;
      uint32_t hiD = __builtin_amdgcn_perm(ro, re, 0x07060302u); // {re.hi16, ro.hi16}
      float le = fe[j] - __uint_as_float(re & 0xFFFF0000u);
      float lo = fo[j] - __uint_as_float(ro & 0xFFFF0000u);
      uint32_t rle = __float_as_uint(le) + 0x8000u;
      uint32_t rlo = __float_as_uint(lo) + 0x8000u;
      uint32_t loD = __builtin_amdgcn_perm(rlo, rle, 0x07060302u);
      int slab = ((mb * 8 + blade) * 2 + kc) * 2;
      Afr[(slab + 0) * 256 + ld * 8 + v] = hiD;
      Afr[(slab + 1) * 256 + ld * 8 + v] = loD;
    }
  }
  __syncthreads();

  // ---- GEMM: accumulate all 8 blades for probe and match -------------------
  const int mb = wv & 1;
  const int nb = wv >> 1;

  v8f accP[8] = {};
  v8f accM[8] = {};

#pragma unroll
  for (int c = 0; c < 4; ++c) {
    const int nBlades = (c == 0 || c == 3) ? 1 : 3;
    const int i0      = (c == 0) ? 0 : (c == 1) ? 1 : (c == 2) ? 4 : 7;
#pragma unroll
    for (int kc = 0; kc < 2; ++kc) {
      // Load B fragments (hi/lo, probe/match) from precomputed workspace.
      auto loadB = [&](int s, int hl) -> v16bf {
        size_t idx = ((((((size_t)s * 4 + c) * 2 + kc) * 4 + nb) * 2 + hl) * 32
                      + lane) * 8;
        const v4u* q = (const v4u*)(wsB + idx);
        v16bf r;
        *(v4u*)&r       = q[0];
        *((v4u*)&r + 1) = q[1];
        return r;
      };
      v16bf BpH = loadB(0, 0), BpL = loadB(0, 1);
      v16bf BmH = loadB(1, 0), BmL = loadB(1, 1);
#pragma unroll
      for (int t = 0; t < nBlades; ++t) {
        const int i = i0 + t;
        const uint32_t* ab = &Afr[(((mb * 8 + i) * 2 + kc) * 2) * 256 + lane * 8];
        v16bf Ah, Al;
        *(v4u*)&Ah       = *(const v4u*)(ab + 0);
        *((v4u*)&Ah + 1) = *(const v4u*)(ab + 4);
        *(v4u*)&Al       = *(const v4u*)(ab + 256);
        *((v4u*)&Al + 1) = *(const v4u*)(ab + 260);
        // fp32 emulation: x*w ~= xh*wh + xh*wl + xl*wh  (fp32 accumulate)
        accP[i] = __builtin_amdgcn_wmma_f32_16x16x32_bf16(false, Ah, false, BpH, (short)0, accP[i], false, false);
        accP[i] = __builtin_amdgcn_wmma_f32_16x16x32_bf16(false, Ah, false, BpL, (short)0, accP[i], false, false);
        accP[i] = __builtin_amdgcn_wmma_f32_16x16x32_bf16(false, Al, false, BpH, (short)0, accP[i], false, false);
        accM[i] = __builtin_amdgcn_wmma_f32_16x16x32_bf16(false, Ah, false, BmH, (short)0, accM[i], false, false);
        accM[i] = __builtin_amdgcn_wmma_f32_16x16x32_bf16(false, Ah, false, BmL, (short)0, accM[i], false, false);
        accM[i] = __builtin_amdgcn_wmma_f32_16x16x32_bf16(false, Al, false, BmH, (short)0, accM[i], false, false);
      }
    }
  }

  // ---- Fused epilogue, packed 2 C-rows per op (V_PK_FMA_F32 path) ----------
  // Accumulator rows (2rp, 2rp+1) are adjacent VGPRs -> legal 64-bit pk pairs.
  const int   n    = nb * 16 + (lane & 15);
  const float bp   = bp_[n];
  const float bm   = bm_[n];
  const long  row0 = rowBase + mb * 16 + (lane >> 4) * 8;
  const float S    = 0.70710678118654752440f;   // 1/sqrt(2)

#pragma unroll
  for (int rp = 0; rp < 4; ++rp) {
    v2f p[8], m[8];
#pragma unroll
    for (int i = 0; i < 8; ++i) {
      p[i] = (v2f){accP[i][2 * rp], accP[i][2 * rp + 1]};
      m[i] = (v2f){accM[i][2 * rp], accM[i][2 * rp + 1]};
    }
    p[0] += bp;   // bias on scalar blade only
    m[0] += bm;

    // Fold 1/sqrt(2) into p: gate sign is unchanged, both select arms scale.
    v2f ps[8];
#pragma unroll
    for (int i = 0; i < 8; ++i) ps[i] = p[i] * S;

    // BETA*BILINEAR == 1  =>  gate b = <p, m> (scaled by S > 0)
    v2f bd = ps[0] * m[0];
#pragma unroll
    for (int i = 1; i < 8; ++i) bd += ps[i] * m[i];
    // q = m0^2+m1^2+m2^2+m3^2 - m4^2-m5^2-m6^2-m7^2
    v2f q = m[0] * m[0] + m[1] * m[1] + m[2] * m[2] + m[3] * m[3]
          - m[4] * m[4] - m[5] * m[5] - m[6] * m[6] - m[7] * m[7];
    v2f qq = q * q + 1e-16f;
    v2f inv;
    inv.x = rsqrtf(sqrtf(qq.x));    // (q^2+eps)^(-1/4)
    inv.y = rsqrtf(sqrtf(qq.y));
    v2f nn[8];
#pragma unroll
    for (int i = 0; i < 8; ++i) nn[i] = m[i] * inv;

    // Cl(3) geometric product ps * nn, blade order [1,e1,e2,e3,e12,e13,e23,e123]
    v2f r[8];
    r[0] = ps[0]*nn[0] + ps[1]*nn[1] + ps[2]*nn[2] + ps[3]*nn[3]
         - ps[4]*nn[4] - ps[5]*nn[5] - ps[6]*nn[6] - ps[7]*nn[7];
    r[1] = ps[0]*nn[1] + ps[1]*nn[0] - ps[2]*nn[4] - ps[3]*nn[5]
         + ps[4]*nn[2] + ps[5]*nn[3] - ps[6]*nn[7] - ps[7]*nn[6];
    r[2] = ps[0]*nn[2] + ps[1]*nn[4] + ps[2]*nn[0] - ps[3]*nn[6]
         - ps[4]*nn[1] + ps[5]*nn[7] + ps[6]*nn[3] + ps[7]*nn[5];
    r[3] = ps[0]*nn[3] + ps[1]*nn[5] + ps[2]*nn[6] + ps[3]*nn[0]
         - ps[4]*nn[7] - ps[5]*nn[1] - ps[6]*nn[2] - ps[7]*nn[4];
    r[4] = ps[0]*nn[4] + ps[1]*nn[2] - ps[2]*nn[1] + ps[3]*nn[7]
         + ps[4]*nn[0] - ps[5]*nn[6] + ps[6]*nn[5] + ps[7]*nn[3];
    r[5] = ps[0]*nn[5] + ps[1]*nn[3] - ps[2]*nn[7] - ps[3]*nn[1]
         + ps[4]*nn[6] + ps[5]*nn[0] - ps[6]*nn[4] - ps[7]*nn[2];
    r[6] = ps[0]*nn[6] + ps[1]*nn[7] + ps[2]*nn[3] - ps[3]*nn[2]
         - ps[4]*nn[5] + ps[5]*nn[4] + ps[6]*nn[0] + ps[7]*nn[1];
    r[7] = ps[0]*nn[7] + ps[1]*nn[6] - ps[2]*nn[5] + ps[3]*nn[4]
         + ps[4]*nn[3] - ps[5]*nn[2] + ps[6]*nn[1] + ps[7]*nn[0];

#pragma unroll
    for (int s2 = 0; s2 < 2; ++s2) {
      bool gate = bd[s2] > 0.0f;
      v4f o0 = { gate ? ps[0][s2] : r[0][s2], gate ? ps[1][s2] : r[1][s2],
                 gate ? ps[2][s2] : r[2][s2], gate ? ps[3][s2] : r[3][s2] };
      v4f o1 = { gate ? ps[4][s2] : r[4][s2], gate ? ps[5][s2] : r[5][s2],
                 gate ? ps[6][s2] : r[6][s2], gate ? ps[7][s2] : r[7][s2] };
      float* op = out + (row0 + 2 * rp + s2) * 512 + n * 8;  // out[b][n][0..7]
      __builtin_nontemporal_store(o0, (v4f*)op);
      __builtin_nontemporal_store(o1, (v4f*)(op + 4));
    }
  }
}

// ---------------------------------------------------------------------------
extern "C" void kernel_launch(void* const* d_in, const int* in_sizes, int n_in,
                              void* d_out, int out_size, void* d_ws, size_t ws_size,
                              hipStream_t stream) {
  (void)n_in; (void)out_size; (void)ws_size;   // ws needs >= 128 KB
  const float* x  = (const float*)d_in[0];
  const float* wp = (const float*)d_in[1];
  const float* bp = (const float*)d_in[2];
  const float* wm = (const float*)d_in[3];
  const float* bm = (const float*)d_in[4];
  float*     outp = (float*)d_out;
  uint32_t*  ws   = (uint32_t*)d_ws;

  // Weight fragments: 32768 threads, one packed dword each.
  clifford_prep_w<<<128, 256, 0, stream>>>(wp, wm, ws);

  const int B = in_sizes[0] / 512;              // (B, 64, 8) fp32
  clifford_main<<<B / 32, 256, 0, stream>>>(x, ws, bp, bm, outp);
}
